// MultiHeadDilatedState_16329465659999
// MI455X (gfx1250) — matrix operand order repaired and checked by reading, not compiled, not run on hardware
//
#include <hip/hip_runtime.h>
#include <hip/hip_bf16.h>
#include <stdint.h>

// ---------------------------------------------------------------------------
// MI455X (gfx1250) implementation: bf16 WMMA GEMMs + fused epilogues.
// ---------------------------------------------------------------------------

typedef __attribute__((ext_vector_type(16))) __bf16 v16bf;
typedef __attribute__((ext_vector_type(8)))  float  v8f;

#define HID   768
#define ROWS  32768        // B*S = 8*4096
#define SEQ   4096
#define NPAIR 384          // 768 bf16 channels as 384 dword pairs

__constant__ int DIL[12][3] = {
  {1,2,4},{1,1,1},{4,8,16},{8,16,32},{32,64,128},{64,128,256},
  {256,512,1024},{1,100,200},{1,500,1000},{1,1024,2048},{3,9,27},{5,25,125}
};

// ----------------------------- helpers -------------------------------------
__device__ __forceinline__ unsigned short f2bfu(float f){
  union{float f; unsigned u;} v; v.f=f;
  unsigned r = v.u + 0x7FFFu + ((v.u>>16)&1u);     // round-to-nearest-even
  return (unsigned short)(r>>16);
}
__device__ __forceinline__ float bflo(unsigned p){ union{unsigned u; float f;} v; v.u = p<<16;          return v.f; }
__device__ __forceinline__ float bfhi(unsigned p){ union{unsigned u; float f;} v; v.u = p & 0xFFFF0000u; return v.f; }
__device__ __forceinline__ float bf2f(unsigned short s){ union{unsigned u; float f;} v; v.u = ((unsigned)s)<<16; return v.f; }
// fast sigmoid: v_exp_f32 + v_rcp_f32 (avoid IEEE div_scale/fma chain)
__device__ __forceinline__ float sigmoidf(float x){
  return __builtin_amdgcn_rcpf(1.f + __expf(-x));
}

__device__ __forceinline__ v8f wmma_bf16(v16bf a, v16bf b, v8f c){
  // v_wmma_f32_16x16x32_bf16  D = A(16x32) * B(32x16) + C
  return __builtin_amdgcn_wmma_f32_16x16x32_bf16(false, a, false, b, (short)0, c, false, false);
}

// A fragment 16x32 (M x K), bf16 row-major, K-contiguous source.
// ISA layout: lanes 0-15 hold M=lane, K {0..7,16..23}; lanes 16-31 same M, K {8..15,24..31}.
__device__ __forceinline__ v16bf load_a_bf16(const unsigned short* A, long long mbase, int ld, int k0, int lane){
  long long row = mbase + (lane & 15);
  const unsigned short* p = A + row*(long long)ld + (k0 + ((lane>>4)<<3));
  union { v16bf v; uint4 q[2]; } u;
  u.q[0] = *(const uint4*)(p);        // K +0..7
  u.q[1] = *(const uint4*)(p+16);     // K +16..23
  return u.v;
}

// B fragment 32x16 (K x N) where B[k][n] = W[n][k], W row-major [N,K] bf16.
// Lanes 0-15 hold col n=lane, K {0..7,16..23}; lanes 16-31 hold same cols, K {8..15,24..31}.
__device__ __forceinline__ v16bf load_b_bf16(const unsigned short* W, int nbase, int ld, int k0, int lane){
  int nrow = nbase + (lane & 15);
  const unsigned short* p = W + (long long)nrow*(long long)ld + (k0 + ((lane>>4)<<3));
  union { v16bf v; uint4 q[2]; } u;
  u.q[0] = *(const uint4*)(p);
  u.q[1] = *(const uint4*)(p+16);
  return u.v;
}

// ----------------------------- kernels -------------------------------------

// vectorized f32 -> bf16 conversion (n must be a multiple of 4)
__global__ __launch_bounds__(256) void cvt4_kernel(const float* __restrict__ in,
                                                   unsigned short* __restrict__ out, int n4){
  int i = blockIdx.x*256 + threadIdx.x;
  if (i < n4){
    float4 f = ((const float4*)in)[i];
    uint2 o;
    o.x = (unsigned)f2bfu(f.x) | ((unsigned)f2bfu(f.y) << 16);
    o.y = (unsigned)f2bfu(f.z) | ((unsigned)f2bfu(f.w) << 16);
    ((uint2*)out)[i] = o;
  }
}

// head_weights = sigmoid(x @ router_w^T + router_b)   [ROWS,12] f32
__global__ __launch_bounds__(256) void router_kernel(const float* __restrict__ x,
                                                     const float* __restrict__ rw,
                                                     const float* __restrict__ rb,
                                                     float* __restrict__ hw){
  __shared__ float w[12*HID];
  for (int i = threadIdx.x; i < 12*HID; i += 256) w[i] = rw[i];
  __syncthreads();
  int lane = threadIdx.x & 31, wid = threadIdx.x >> 5;
  long long row = (long long)blockIdx.x*8 + wid;
  const float* xp = x + row*HID;
  float s[12];
#pragma unroll
  for (int h = 0; h < 12; ++h) s[h] = 0.f;
  for (int kk = lane; kk < HID; kk += 32){
    float xv = xp[kk];
#pragma unroll
    for (int h = 0; h < 12; ++h) s[h] += xv * w[h*HID + kk];
  }
#pragma unroll
  for (int h = 0; h < 12; ++h)
#pragma unroll
    for (int off = 16; off; off >>= 1) s[h] += __shfl_xor(s[h], off, 32);
  if (lane == 0){
#pragma unroll
    for (int h = 0; h < 12; ++h) hw[row*12 + h] = sigmoidf(s[h] + rb[h]);
  }
}

// Gate GEMM: xg = (xb @ Wa^T) * sigmoid(xb @ Wb^T), Wa/Wb = halves of gate_w (bf16).
// Wave tile: 32(M) x 32(N) gated output, dual accumulators per N tile.
// Block: 8 waves as 4(M) x 2(N) -> 128 x 64.  Grid: (N=12 fastest, M=256).
__global__ __launch_bounds__(256) void gemm_gate(const unsigned short* __restrict__ xb,
                                                 const unsigned short* __restrict__ wg,
                                                 unsigned short* __restrict__ xg){
  int lane = threadIdx.x & 31, wid = threadIdx.x >> 5;
  int wm = wid >> 1, wn = wid & 1;
  long long m0 = (long long)blockIdx.y*128 + wm*32;
  int       n0 = blockIdx.x*64 + wn*32;
  v8f zero = {0.f,0.f,0.f,0.f,0.f,0.f,0.f,0.f};
  v8f acc[2][2][2];
#pragma unroll
  for (int mi=0; mi<2; ++mi)
#pragma unroll
    for (int nt=0; nt<2; ++nt){ acc[mi][nt][0]=zero; acc[mi][nt][1]=zero; }

  for (int kt = 0; kt < HID/32; ++kt){
    int k0 = kt*32;
    v16bf a0 = load_a_bf16(xb, m0,      HID, k0, lane);
    v16bf a1 = load_a_bf16(xb, m0 + 16, HID, k0, lane);
#pragma unroll
    for (int nt = 0; nt < 2; ++nt){
      v16bf bA = load_b_bf16(wg, n0 + nt*16,       HID, k0, lane);
      v16bf bB = load_b_bf16(wg, n0 + nt*16 + HID, HID, k0, lane);
      acc[0][nt][0] = wmma_bf16(a0, bA, acc[0][nt][0]);
      acc[0][nt][1] = wmma_bf16(a0, bB, acc[0][nt][1]);
      acc[1][nt][0] = wmma_bf16(a1, bA, acc[1][nt][0]);
      acc[1][nt][1] = wmma_bf16(a1, bB, acc[1][nt][1]);
    }
  }
  int col = lane & 15, rbase = (lane >> 4) * 8;
#pragma unroll
  for (int mi = 0; mi < 2; ++mi)
#pragma unroll
    for (int nt = 0; nt < 2; ++nt)
#pragma unroll
      for (int r = 0; r < 8; ++r){
        long long m = m0 + mi*16 + rbase + r;
        int n = n0 + nt*16 + col;
        float res = acc[mi][nt][0][r] * sigmoidf(acc[mi][nt][1][r]);
        xg[m*HID + n] = f2bfu(res);
      }
}

// One dilated causal depthwise conv stage (bf16 in/out, channel pairs).
// out[s] = in[s] + bias + sum_k w[k]*in[s-(3-k)*d]; stage 2 multiplies head_weights.
__global__ __launch_bounds__(256) void conv_stage(const unsigned short* __restrict__ hin,
                                                  unsigned short* __restrict__ hout,
                                                  const float* __restrict__ conv_w,
                                                  const float* __restrict__ conv_b,
                                                  const float* __restrict__ hw,
                                                  int stage, int mulhw){
  long long idx = (long long)blockIdx.x*256 + threadIdx.x;   // ROWS * NPAIR threads
  int cp = (int)(idx % NPAIR);
  long long bs = idx / NPAIR;
  int s = (int)(bs % SEQ);
  int c = cp*2, head = c >> 6, cc = c & 63;
  int d = DIL[head][stage];
  const float* wp = conv_w + ((head*3 + stage)*64 + cc)*4;   // [k] for cc, +4 for cc+1
  float b0 = conv_b[(head*3 + stage)*64 + cc];
  float b1 = conv_b[(head*3 + stage)*64 + cc + 1];
  const unsigned* in32 = (const unsigned*)hin;
  unsigned selfv = in32[bs*NPAIR + cp];
  float a0 = bflo(selfv) + b0;
  float a1 = bfhi(selfv) + b1;
#pragma unroll
  for (int k = 0; k < 4; ++k){
    int off = (3 - k) * d;
    if (s >= off){
      unsigned t = in32[(bs - off)*NPAIR + cp];
      a0 += wp[k]     * bflo(t);
      a1 += wp[4 + k] * bfhi(t);
    }
  }
  if (mulhw){
    float m = hw[bs*12 + head];
    a0 *= m; a1 *= m;
  }
  ((unsigned*)hout)[bs*NPAIR + cp] =
      (unsigned)f2bfu(a0) | ((unsigned)f2bfu(a1) << 16);
}

// Epilogue-fused bf16 GEMM: wave tile 32(M) x 64(N); block 128 x 128.
// Grid: (N=6 fastest, M=256) so all N-blocks of one A slab run adjacently (L2 reuse).
// EPI==0: out2(bf16) = co * sigmoid(acc + bias)     (mix gate)
// EPI==1: outf(f32)  = acc + bias                   (final mixing)
template<int EPI>
__global__ __launch_bounds__(256) void gemm_epi(const unsigned short* __restrict__ A,
                                                const unsigned short* __restrict__ W,
                                                const float* __restrict__ bias,
                                                const unsigned short* __restrict__ co,
                                                unsigned short* __restrict__ outb,
                                                float* __restrict__ outf){
  int lane = threadIdx.x & 31, wid = threadIdx.x >> 5;
  int wm = wid >> 1, wn = wid & 1;
  long long m0 = (long long)blockIdx.y*128 + wm*32;
  int       n0 = blockIdx.x*128 + wn*64;
  v8f zero = {0.f,0.f,0.f,0.f,0.f,0.f,0.f,0.f};
  v8f acc[2][4];
#pragma unroll
  for (int mi=0; mi<2; ++mi)
#pragma unroll
    for (int nt=0; nt<4; ++nt) acc[mi][nt] = zero;

  for (int kt = 0; kt < HID/32; ++kt){
    int k0 = kt*32;
    v16bf a0 = load_a_bf16(A, m0,      HID, k0, lane);
    v16bf a1 = load_a_bf16(A, m0 + 16, HID, k0, lane);
#pragma unroll
    for (int nt = 0; nt < 4; ++nt){
      v16bf b = load_b_bf16(W, n0 + nt*16, HID, k0, lane);
      acc[0][nt] = wmma_bf16(a0, b, acc[0][nt]);
      acc[1][nt] = wmma_bf16(a1, b, acc[1][nt]);
    }
  }
  int col = lane & 15, rbase = (lane >> 4) * 8;
#pragma unroll
  for (int mi = 0; mi < 2; ++mi)
#pragma unroll
    for (int nt = 0; nt < 4; ++nt)
#pragma unroll
      for (int r = 0; r < 8; ++r){
        long long m = m0 + mi*16 + rbase + r;
        int n = n0 + nt*16 + col;
        float v = acc[mi][nt][r] + bias[n];
        if (EPI == 0){
          float c = bf2f(co[m*HID + n]);
          outb[m*HID + n] = f2bfu(c * sigmoidf(v));
        } else {
          outf[m*HID + n] = v;
        }
      }
}

// ----------------------------- launcher ------------------------------------
extern "C" void kernel_launch(void* const* d_in, const int* in_sizes, int n_in,
                              void* d_out, int out_size, void* d_ws, size_t ws_size,
                              hipStream_t stream) {
  const float* x          = (const float*)d_in[0];   // [8,4096,768]
  const float* gate_w     = (const float*)d_in[1];   // [1536,768]
  const float* conv_w     = (const float*)d_in[2];   // [12,3,64,4]
  const float* conv_b     = (const float*)d_in[3];   // [12,3,64]
  const float* router_w   = (const float*)d_in[4];   // [12,768]
  const float* router_b   = (const float*)d_in[5];   // [12]
  const float* mix_gate_w = (const float*)d_in[6];   // [768,768]
  const float* mix_gate_b = (const float*)d_in[7];   // [768]
  const float* mixing_w   = (const float*)d_in[8];   // [768,768]
  const float* mixing_b   = (const float*)d_in[9];   // [768]
  float* out = (float*)d_out;

  char* ws = (char*)d_ws;
  // workspace layout (bytes, 256-aligned sections)
  unsigned short* wg_b = (unsigned short*)(ws + 0);          //  2,359,296  gate_w bf16
  unsigned short* wm_b = (unsigned short*)(ws + 2359296);    //  1,179,648  mix_gate_w bf16
  unsigned short* wx_b = (unsigned short*)(ws + 3538944);    //  1,179,648  mixing_w bf16
  float*          hwp  = (float*)         (ws + 4718592);    //  1,572,864  head_weights
  unsigned short* xg   = (unsigned short*)(ws + 6291456);    // 50,331,648  x_gated bf16
  unsigned short* ping = (unsigned short*)(ws + 56623104);   // 50,331,648
  unsigned short* pong = (unsigned short*)(ws + 106954752);  // 50,331,648  (end 157,286,400)
  unsigned short* xb   = pong;   // x as bf16; dead before conv stage 1 writes pong

  // 1) weights + x -> bf16 (vectorized)
  cvt4_kernel<<<(1536*HID/4 + 255)/256, 256, 0, stream>>>(gate_w,     wg_b, 1536*HID/4);
  cvt4_kernel<<<(HID*HID/4  + 255)/256, 256, 0, stream>>>(mix_gate_w, wm_b, HID*HID/4);
  cvt4_kernel<<<(HID*HID/4  + 255)/256, 256, 0, stream>>>(mixing_w,   wx_b, HID*HID/4);
  cvt4_kernel<<<(ROWS*HID/4 + 255)/256, 256, 0, stream>>>(x,          xb,   ROWS*HID/4);

  // 2) router head weights
  router_kernel<<<ROWS/8, 256, 0, stream>>>(x, router_w, router_b, hwp);

  // 3) gate GEMM with fused sigmoid gating -> xg (bf16)
  gemm_gate<<<dim3(HID/64, ROWS/128), 256, 0, stream>>>(xb, wg_b, xg);

  // 4) three sequential dilated depthwise conv stages (stage 2 fuses head_weights)
  int conv_blocks = (int)(((long long)ROWS * NPAIR) / 256);  // 49152
  conv_stage<<<conv_blocks, 256, 0, stream>>>(xg,   ping, conv_w, conv_b, hwp, 0, 0);
  conv_stage<<<conv_blocks, 256, 0, stream>>>(ping, pong, conv_w, conv_b, hwp, 1, 0);
  conv_stage<<<conv_blocks, 256, 0, stream>>>(pong, ping, conv_w, conv_b, hwp, 2, 1);

  // 5) mix-gate GEMM with fused out*sigmoid(acc+b) -> pong (bf16)
  gemm_epi<0><<<dim3(HID/128, ROWS/128), 256, 0, stream>>>(ping, wm_b, mix_gate_b, ping, pong, nullptr);

  // 6) final mixing GEMM + bias -> d_out (f32)
  gemm_epi<1><<<dim3(HID/128, ROWS/128), 256, 0, stream>>>(pong, wx_b, mixing_b, nullptr, nullptr, out);
}
